// Gtu_3487513444805
// MI455X (gfx1250) — compile-verified, matrix operand order checked by reading, not run on hardware
//
#include <hip/hip_runtime.h>
#include <hip/hip_bf16.h>
#include <math.h>

typedef __attribute__((ext_vector_type(16))) __bf16 bf16x16;
typedef __attribute__((ext_vector_type(8)))  __bf16 bf16x8;
typedef __attribute__((ext_vector_type(8)))  float  f32x8;
typedef __attribute__((ext_vector_type(2)))  float  f32x2;

#define NSEQ 2048
#define DMODEL 1024
#define BATCH 4

// ---------------------------------------------------------------------------
// Stage 0: RPE MLP -> decayed Toeplitz table T[(2N-1) x D]
// One wave32 per relative position k in [-(N-1), N-1]; lane = feature (F=32).
// ---------------------------------------------------------------------------
__global__ __launch_bounds__(256) void rpe_kernel(
    const float* __restrict__ pos_w,   // (1,32)
    const float* __restrict__ ln_w,    // (3,32)
    const float* __restrict__ ln_b,    // (3,32)
    const float* __restrict__ fc_w,    // (3,32,32)
    const float* __restrict__ oln_w,   // (32,)
    const float* __restrict__ oln_b,   // (32,)
    const float* __restrict__ out_w,   // (32,1024)
    float* __restrict__ T)
{
    const int wave = threadIdx.x >> 5;
    const int lane = threadIdx.x & 31;
    const int p = blockIdx.x * 8 + wave;          // row of T
    const int total = 2 * NSEQ - 1;
    if (p >= total) return;

    const float kf = (float)(p - (NSEQ - 1));     // signed relative position
    float h = kf * pos_w[lane];

    #pragma unroll
    for (int L = 0; L < 3; ++L) {
        float mu = h;
        for (int o = 16; o > 0; o >>= 1) mu += __shfl_xor(mu, o, 32);
        mu *= (1.0f / 32.0f);
        float dv = h - mu;
        float var = dv * dv;
        for (int o = 16; o > 0; o >>= 1) var += __shfl_xor(var, o, 32);
        var *= (1.0f / 32.0f);
        float xn = dv * rsqrtf(var + 1e-5f) * ln_w[L * 32 + lane] + ln_b[L * 32 + lane];
        float s = xn / (1.0f + __expf(-xn));      // silu
        float nh = 0.0f;
        #pragma unroll
        for (int f = 0; f < 32; ++f)
            nh += __shfl(s, f, 32) * fc_w[L * 1024 + f * 32 + lane];
        h = nh;
    }
    // output LN + silu
    float mu = h;
    for (int o = 16; o > 0; o >>= 1) mu += __shfl_xor(mu, o, 32);
    mu *= (1.0f / 32.0f);
    float dv = h - mu;
    float var = dv * dv;
    for (int o = 16; o > 0; o >>= 1) var += __shfl_xor(var, o, 32);
    var *= (1.0f / 32.0f);
    float xn = dv * rsqrtf(var + 1e-5f) * oln_w[lane] + oln_b[lane];
    float g = xn / (1.0f + __expf(-xn));

    // decay gamma^|k|, gamma = 0.99  (ln 0.99 = -0.01005033585)
    const float decay = __expf(fabsf(kf) * -0.010050335853501441f);

    // project 32 -> 1024 and store row p
    #pragma unroll 4
    for (int q = 0; q < 32; ++q) {
        int d = q * 32 + lane;
        float acc = 0.0f;
        #pragma unroll
        for (int f = 0; f < 32; ++f)
            acc += __shfl(g, f, 32) * out_w[f * DMODEL + d];
        T[(size_t)p * DMODEL + d] = acc * decay;
    }
}

// ---------------------------------------------------------------------------
// Stage 1: uv = silu(x @ uv_w), split -> U, V     (bf16 WMMA, f32 accumulate)
// Block tile 128x64, 8 waves in 4x2 grid, each wave a 32x32 region.
// A staged row-major (vectorized b128), B staged TRANSPOSED so both fragment
// gathers are two aligned ds_load_b128 per fragment. Rows padded +8 bf16
// (16 B) for bank rotation while keeping 16-B alignment.
// ---------------------------------------------------------------------------
__global__ __launch_bounds__(256) void gemm1_silu(
    const float* __restrict__ X,       // (M, K)
    const float* __restrict__ W,       // (K, NN)
    float* __restrict__ U,             // (M, 1024)
    float* __restrict__ V,             // (M, 1024)
    int M, int NN, int K)
{
    __shared__ __bf16 As[128][40];     // 10 KB, row stride 80 B
    __shared__ __bf16 Bt[64][40];      // 5 KB,  Bt[n][k]

    const int tid = threadIdx.x;
    const int wave = tid >> 5, lane = tid & 31;
    const int wm = wave >> 1, wn = wave & 1;      // 4x2 wave grid
    const int m0 = blockIdx.x * 128, n0 = blockIdx.y * 64;
    const int r = lane & 15, hi = lane >> 4;

    // staging coordinates
    const int arow_st = tid >> 1;            // 0..127
    const int ahalf   = tid & 1;             // 16-float half-row
    const int bn_st   = tid & 63;            // 0..63  (output column)
    const int bk_st   = (tid >> 6) * 8;      // 0,8,16,24 (k segment)

    f32x8 acc[2][2] = {};

    for (int k0 = 0; k0 < K; k0 += 32) {
        // ---- stage A tile: contiguous 16 floats/thread -> bf16 ----
        {
            const float4* src = reinterpret_cast<const float4*>(
                &X[(size_t)(m0 + arow_st) * K + k0 + ahalf * 16]);
            float4 f0 = src[0], f1 = src[1], f2 = src[2], f3 = src[3];
            bf16x8 lo, hi8;
            lo[0] = (__bf16)f0.x; lo[1] = (__bf16)f0.y; lo[2] = (__bf16)f0.z; lo[3] = (__bf16)f0.w;
            lo[4] = (__bf16)f1.x; lo[5] = (__bf16)f1.y; lo[6] = (__bf16)f1.z; lo[7] = (__bf16)f1.w;
            hi8[0] = (__bf16)f2.x; hi8[1] = (__bf16)f2.y; hi8[2] = (__bf16)f2.z; hi8[3] = (__bf16)f2.w;
            hi8[4] = (__bf16)f3.x; hi8[5] = (__bf16)f3.y; hi8[6] = (__bf16)f3.z; hi8[7] = (__bf16)f3.w;
            *(bf16x8*)&As[arow_st][ahalf * 16]     = lo;
            *(bf16x8*)&As[arow_st][ahalf * 16 + 8] = hi8;
        }
        // ---- stage B tile transposed: lane-coalesced strided loads ----
        {
            const float* src = &W[(size_t)(k0 + bk_st) * NN + n0 + bn_st];
            bf16x8 col;
            #pragma unroll
            for (int q = 0; q < 8; ++q)
                col[q] = (__bf16)src[(size_t)q * NN];
            *(bf16x8*)&Bt[bn_st][bk_st] = col;
        }
        if (k0 + 32 < K)  // pull next A tile toward L2 (global_prefetch_b8)
            __builtin_prefetch(&X[(size_t)(m0 + arow_st) * K + k0 + 32], 0, 1);
        __syncthreads();

        // ---- fragments: two aligned ds_load_b128 each ----
        bf16x16 afr[2], bfr[2];
        #pragma unroll
        for (int ti = 0; ti < 2; ++ti) {
            int arow = wm * 32 + ti * 16 + r;
            bf16x8 a0 = *(const bf16x8*)&As[arow][hi * 8];
            bf16x8 a1 = *(const bf16x8*)&As[arow][16 + hi * 8];
            afr[ti] = __builtin_shufflevector(a0, a1, 0,1,2,3,4,5,6,7,8,9,10,11,12,13,14,15);
        }
        #pragma unroll
        for (int tj = 0; tj < 2; ++tj) {
            int bcol = wn * 32 + tj * 16 + r;
            bf16x8 b0 = *(const bf16x8*)&Bt[bcol][hi * 8];
            bf16x8 b1 = *(const bf16x8*)&Bt[bcol][16 + hi * 8];
            bfr[tj] = __builtin_shufflevector(b0, b1, 0,1,2,3,4,5,6,7,8,9,10,11,12,13,14,15);
        }
        #pragma unroll
        for (int ti = 0; ti < 2; ++ti)
            #pragma unroll
            for (int tj = 0; tj < 2; ++tj)
                acc[ti][tj] = __builtin_amdgcn_wmma_f32_16x16x32_bf16(
                    false, afr[ti], false, bfr[tj], (short)0, acc[ti][tj], false, false);
        __syncthreads();
    }

    // epilogue: whole 64-wide n-tile lands entirely in U or in V
    float* dst = (n0 < DMODEL) ? U : V;
    const int cbase = (n0 < DMODEL) ? n0 : (n0 - DMODEL);
    #pragma unroll
    for (int ti = 0; ti < 2; ++ti)
        #pragma unroll
        for (int tj = 0; tj < 2; ++tj)
            #pragma unroll
            for (int e = 0; e < 8; ++e) {
                int row = m0 + wm * 32 + ti * 16 + e + hi * 8;
                int col = cbase + wn * 32 + tj * 16 + r;
                float v = acc[ti][tj][e];
                v = v / (1.0f + __expf(-v));
                dst[(size_t)row * DMODEL + col] = v;
            }
}

// ---------------------------------------------------------------------------
// Stage 2: y[b,i,d] = sum_j T[i-j+N-1, d] * v[b,j,d]   (depthwise Toeplitz)
// Tiles: 128 i x 64 d per block, loop j in tiles of 128.
// BOTH tiles staged via CDNA5 async global->LDS b128 copies, one asynccnt wait.
// ---------------------------------------------------------------------------
__global__ __launch_bounds__(256) void toeplitz_kernel(
    const float* __restrict__ T,       // (2N-1, D)
    const float* __restrict__ V,       // (B*N, D)
    float* __restrict__ Y)             // (B*N, D)
{
    __shared__ float vs[128][64];      // 32 KB
    __shared__ float ts[255][64];      // 63.75 KB

    const int b  = blockIdx.z;
    const int i0 = blockIdx.x * 128;
    const int d0 = blockIdx.y * 64;
    const int tid = threadIdx.x;
    const int dl = tid & 63;
    const int ib = tid >> 6;           // 0..3 -> rows ib + 4*r

    float acc[32] = {};

    for (int j0 = 0; j0 < NSEQ; j0 += 128) {
        // ---- async copy v tile: 128 rows x 64 cols ----
        {
            int jr = tid >> 1;                         // 0..127
            int hh = tid & 1;                          // half-row (32 floats)
            unsigned lds_a = (unsigned)(size_t)&vs[jr][hh * 32];
            unsigned goff  = (unsigned)((((size_t)b * NSEQ + (j0 + jr)) * DMODEL
                                         + d0 + hh * 32) * 4);
            #pragma unroll
            for (int c = 0; c < 8; ++c) {
                asm volatile("global_load_async_to_lds_b128 %0, %1, %2"
                             :: "v"(lds_a + c * 16), "v"(goff + c * 16), "s"(V)
                             : "memory");
            }
        }
        // ---- async copy T tile: 255 rows x 64 cols, 16-B chunks ----
        {
            int m0 = i0 - j0 - 127 + (NSEQ - 1);       // always in [0, 2N-2-254]
            for (int c = tid; c < 255 * 16; c += 256) {
                int row = c >> 4, part = c & 15;
                unsigned lds_a = (unsigned)(size_t)&ts[row][part * 4];
                unsigned goff  = (unsigned)(((size_t)(m0 + row) * DMODEL
                                             + d0 + part * 4) * 4);
                asm volatile("global_load_async_to_lds_b128 %0, %1, %2"
                             :: "v"(lds_a), "v"(goff), "s"(T)
                             : "memory");
            }
        }
        asm volatile("s_wait_asynccnt 0" ::: "memory");
        __syncthreads();

        // ---- accumulate: reuse vs[j][dl] across 32 output rows ----
        for (int j = 0; j < 128; ++j) {
            float vv = vs[j][dl];
            #pragma unroll
            for (int rr = 0; rr < 32; ++rr) {
                int il = ib + rr * 4;
                acc[rr] += ts[il - j + 127][dl] * vv;
            }
        }
        __syncthreads();
    }

    #pragma unroll
    for (int rr = 0; rr < 32; ++rr) {
        int il = ib + rr * 4;
        Y[((size_t)b * NSEQ + i0 + il) * DMODEL + d0 + dl] = acc[rr];
    }
}

// ---------------------------------------------------------------------------
// Stage 3: out = (U .* Y) @ o_w   using native fp32 WMMA 16x16x4.
// Block tile 128x64, 8 waves (4x2), each wave 32x32; K-chunk 32 = 8 sub-steps.
// A and transposed-B tiles padded to 36 floats/row: float4 staging stays
// 16-B aligned, fragment ds_load_b64 conflict-free.
// ---------------------------------------------------------------------------
__global__ __launch_bounds__(256) void gemm2_f32(
    const float* __restrict__ U,       // (M, K)
    const float* __restrict__ Y,       // (M, K)
    const float* __restrict__ Ow,      // (K, N)
    float* __restrict__ Out,           // (M, N)
    int M, int N, int K)
{
    __shared__ float As[128][36];      // 18.4 KB
    __shared__ float Bt[64][36];       // 9.2 KB, Bt[n][k]

    const int tid = threadIdx.x;
    const int wave = tid >> 5, lane = tid & 31;
    const int wm = wave >> 1, wn = wave & 1;
    const int m0 = blockIdx.x * 128, n0 = blockIdx.y * 64;
    const int r = lane & 15, hi = lane >> 4;

    const int arow_st = tid >> 1;
    const int ahalf   = tid & 1;
    const int bn_st   = tid & 63;
    const int bk_st   = (tid >> 6) * 8;

    f32x8 acc[2][2] = {};

    for (int k0 = 0; k0 < K; k0 += 32) {
        // ---- A tile = U .* Y (fused gating), 16 floats/thread ----
        {
            size_t g = (size_t)(m0 + arow_st) * K + k0 + ahalf * 16;
            const float4* su = reinterpret_cast<const float4*>(&U[g]);
            const float4* sy = reinterpret_cast<const float4*>(&Y[g]);
            float4* dst = reinterpret_cast<float4*>(&As[arow_st][ahalf * 16]);
            #pragma unroll
            for (int q = 0; q < 4; ++q) {
                float4 a = su[q], b = sy[q];
                float4 p;
                p.x = a.x * b.x; p.y = a.y * b.y; p.z = a.z * b.z; p.w = a.w * b.w;
                dst[q] = p;
            }
        }
        // ---- B tile transposed: lane-coalesced strided loads ----
        {
            const float* src = &Ow[(size_t)(k0 + bk_st) * N + n0 + bn_st];
            float c[8];
            #pragma unroll
            for (int q = 0; q < 8; ++q)
                c[q] = src[(size_t)q * N];
            float4* dst = reinterpret_cast<float4*>(&Bt[bn_st][bk_st]);
            dst[0] = make_float4(c[0], c[1], c[2], c[3]);
            dst[1] = make_float4(c[4], c[5], c[6], c[7]);
        }
        __syncthreads();

        #pragma unroll
        for (int kk = 0; kk < 8; ++kk) {           // 8 sub-steps of K=4
            f32x2 afr[2], bfr[2];
            #pragma unroll
            for (int ti = 0; ti < 2; ++ti) {
                int arow = wm * 32 + ti * 16 + r;
                afr[ti] = *(const f32x2*)&As[arow][kk * 4 + hi * 2];
            }
            #pragma unroll
            for (int tj = 0; tj < 2; ++tj) {
                int bcol = wn * 32 + tj * 16 + r;
                bfr[tj] = *(const f32x2*)&Bt[bcol][kk * 4 + hi * 2];
            }
            #pragma unroll
            for (int ti = 0; ti < 2; ++ti)
                #pragma unroll
                for (int tj = 0; tj < 2; ++tj)
                    acc[ti][tj] = __builtin_amdgcn_wmma_f32_16x16x4_f32(
                        false, afr[ti], false, bfr[tj], (short)0, acc[ti][tj], false, false);
        }
        __syncthreads();
    }

    #pragma unroll
    for (int ti = 0; ti < 2; ++ti)
        #pragma unroll
        for (int tj = 0; tj < 2; ++tj)
            #pragma unroll
            for (int e = 0; e < 8; ++e) {
                int row = m0 + wm * 32 + ti * 16 + e + hi * 8;
                int col = n0 + wn * 32 + tj * 16 + r;
                Out[(size_t)row * N + col] = acc[ti][tj][e];
            }
}

// ---------------------------------------------------------------------------
extern "C" void kernel_launch(void* const* d_in, const int* in_sizes, int n_in,
                              void* d_out, int out_size, void* d_ws, size_t ws_size,
                              hipStream_t stream) {
    const float* x         = (const float*)d_in[0];
    const float* uv_w      = (const float*)d_in[1];
    const float* o_w       = (const float*)d_in[2];
    const float* rpe_pos_w = (const float*)d_in[3];
    const float* rpe_ln_w  = (const float*)d_in[4];
    const float* rpe_ln_b  = (const float*)d_in[5];
    const float* rpe_fc_w  = (const float*)d_in[6];
    const float* rpe_oln_w = (const float*)d_in[7];
    const float* rpe_oln_b = (const float*)d_in[8];
    const float* rpe_out_w = (const float*)d_in[9];
    float* out = (float*)d_out;

    const int M = BATCH * NSEQ;                 // 8192
    float* ws = (float*)d_ws;
    float* T = ws;                              // (2N-1) * D
    size_t TSZ = (size_t)(2 * NSEQ - 1) * DMODEL;
    float* U = T + TSZ;                         // M * D
    float* V = U + (size_t)M * DMODEL;          // M * D
    float* Y = V + (size_t)M * DMODEL;          // M * D

    // Stage 0: Toeplitz coefficient table
    rpe_kernel<<<(2 * NSEQ - 1 + 7) / 8, 256, 0, stream>>>(
        rpe_pos_w, rpe_ln_w, rpe_ln_b, rpe_fc_w, rpe_oln_w, rpe_oln_b, rpe_out_w, T);

    // Stage 1: uv = silu(x @ uv_w) -> U, V
    gemm1_silu<<<dim3(M / 128, (2 * DMODEL) / 64), 256, 0, stream>>>(
        x, uv_w, U, V, M, 2 * DMODEL, DMODEL);

    // Stage 2: Y = toeplitz(V)
    toeplitz_kernel<<<dim3(NSEQ / 128, DMODEL / 64, BATCH), 256, 0, stream>>>(T, V, Y);

    // Stage 3: out = (U .* Y) @ o_w
    gemm2_f32<<<dim3(M / 128, DMODEL / 64), 256, 0, stream>>>(
        U, Y, o_w, out, M, DMODEL, DMODEL);
}
